// NeuronMemory_8512625180871
// MI455X (gfx1250) — matrix operand order repaired and checked by reading, not compiled, not run on hardware
//
#include <hip/hip_runtime.h>
#include <hip/hip_bf16.h>
#include <math.h>

#define D_MODEL      1024
#define N_KNOW       32768
#define K_RANK       128
#define COARSE_K     20
#define FINE_K       10
#define TOKENS_TOTAL 8192
#define TOK_TILE     16
#define NWAVES       8
#define NTHREADS     256
#define KSTEPS       (D_MODEL / 32)   // 32 WMMA K-steps
#define TOKENS_BLK   64               // tokens per coarse block (4 M tiles)
#define MT           4                // M tiles per coarse block
#define NSPLIT       4                // N-dimension splits (grid = 128*4 = 512)

typedef __attribute__((ext_vector_type(16))) __bf16          bf16x16;
typedef __attribute__((ext_vector_type(16))) unsigned short  u16x16;
typedef __attribute__((ext_vector_type(8)))  unsigned short  u16x8;
typedef __attribute__((ext_vector_type(8)))  float           f32x8;

// round-to-nearest-even f32 -> bf16 bits
__device__ __forceinline__ unsigned short f2bf(float f) {
    unsigned int u = __builtin_bit_cast(unsigned int, f);
    u += 0x7FFFu + ((u >> 16) & 1u);
    return (unsigned short)(u >> 16);
}

// ---------------------------------------------------------------------------
// Pack [D_MODEL x N] f32 weights -> bf16 in WMMA B-fragment order:
//   out[((nt*KSTEPS + s)*32 + lane)*16 + j] = bf16( W[(s*32 + (lane>>4)*16 + j)*N + nt*16 + (lane&15)] )
// One thread produces one 32-byte fragment slice.
// ---------------------------------------------------------------------------
__global__ __launch_bounds__(NTHREADS)
void pack_w_kernel(const float* __restrict__ W, unsigned short* __restrict__ out,
                   int N) {
    const size_t g    = (size_t)blockIdx.x * NTHREADS + threadIdx.x;
    const int    lane = (int)(g & 31);
    const int    s    = (int)((g >> 5) & (KSTEPS - 1));
    const size_t nt   = g >> 10;
    const int    n     = (int)(nt * 16 + (lane & 15));
    const int    kbase = s * 32 + ((lane >> 4) << 4);
    u16x16 v;
    #pragma unroll
    for (int j = 0; j < 16; ++j)
        v[j] = f2bf(W[(size_t)(kbase + j) * N + n]);
    *(u16x16*)(out + g * 16) = v;
}

// x f32 -> flat bf16 (row-major), 8 elements/thread
__global__ __launch_bounds__(NTHREADS)
void pack_x_kernel(const float* __restrict__ x, unsigned short* __restrict__ out) {
    const size_t g = ((size_t)blockIdx.x * NTHREADS + threadIdx.x) * 8;
    u16x8 v;
    #pragma unroll
    for (int j = 0; j < 8; ++j) v[j] = f2bf(x[g + j]);
    *(u16x8*)(out + g) = v;
}

// ---------------------------------------------------------------------------
// Kernel: coarse routing partials. Block = 64 tokens x (8192 columns of one
// N-split). 8 waves; each wave: 1 B fragment reused across 4 M tiles
// (4 WMMA / step). Per-wave sorted top-20 lists, merged per split at the end.
// ---------------------------------------------------------------------------
__global__ __launch_bounds__(NTHREADS)
void coarse_partial_kernel(const unsigned short* __restrict__ xb,
                           const unsigned short* __restrict__ wrp,
                           float* __restrict__ pval, int* __restrict__ pidx) {
    __shared__ __align__(16) unsigned short sx[TOKENS_BLK][D_MODEL]; // 128 KB
    __shared__ float          tv[NWAVES][TOKENS_BLK][COARSE_K];      // 40 KB
    __shared__ unsigned short ti[NWAVES][TOKENS_BLK][COARSE_K];      // 20 KB
    __shared__ float          stage[NWAVES][TOK_TILE][TOK_TILE];     // 8 KB

    const int tb   = blockIdx.x >> 2;   // token block 0..127
    const int sp   = blockIdx.x & 3;    // N split 0..3
    const int t0   = tb * TOKENS_BLK;
    const int tid  = threadIdx.x;
    const int wave = tid >> 5;
    const int lane = tid & 31;
    const int mrow = lane & 15;
    const int half = lane >> 4;

    // fill x tile: pure b128 copies of pre-converted bf16
    {
        const u16x8* src = (const u16x8*)xb + (size_t)t0 * (D_MODEL / 8);
        u16x8* dst = (u16x8*)&sx[0][0];
        for (int i = tid; i < TOKENS_BLK * (D_MODEL / 8); i += NTHREADS) dst[i] = src[i];
    }
    for (int i = tid; i < NWAVES * TOKENS_BLK * COARSE_K; i += NTHREADS) {
        (&tv[0][0][0])[i] = -3.0e38f;
        (&ti[0][0][0])[i] = 0;
    }
    __syncthreads();

    const int TILES_SPLIT = (N_KNOW / 16) / NSPLIT;      // 512 tiles per split
    const int ITERS       = TILES_SPLIT / NWAVES;        // 64 per wave
    for (int it = 0; it < ITERS; ++it) {
        const int nt = sp * TILES_SPLIT + it * NWAVES + wave;
        const int n0 = nt * 16;
        const u16x16* bp = (const u16x16*)wrp + (size_t)nt * (KSTEPS * 32);

        f32x8 cacc[MT] = {};
        for (int s = 0; s < KSTEPS; ++s) {
            // one coalesced 32B B-fragment load per lane
            u16x16 bv = bp[s * 32 + lane];
            bf16x16 b = __builtin_bit_cast(bf16x16, bv);
            #pragma unroll
            for (int mt = 0; mt < MT; ++mt) {
                const u16x8* row = (const u16x8*)&sx[mt * 16 + mrow][0];
                u16x8 lo = row[(s << 2) + (half << 1)];       // K = s*32 + half*16 ..
                u16x8 hi = row[(s << 2) + (half << 1) + 1];   // K = s*32 + half*16 + 8 ..
                u16x16 av;
                #pragma unroll
                for (int j = 0; j < 8; ++j) { av[j] = lo[j]; av[j + 8] = hi[j]; }
                bf16x16 a = __builtin_bit_cast(bf16x16, av);
                cacc[mt] = __builtin_amdgcn_wmma_f32_16x16x32_bf16(
                               false, a, false, b, (short)0, cacc[mt], false, false);
            }
        }

        // stage each 16x16 logit tile, maintain per-wave sorted top-20
        #pragma unroll
        for (int mt = 0; mt < MT; ++mt) {
            #pragma unroll
            for (int r = 0; r < 8; ++r)
                stage[wave][(half << 3) + r][mrow] = cacc[mt][r];
            __syncthreads();
            if (lane < TOK_TILE) {
                float*          tvr = &tv[wave][mt * 16 + lane][0];
                unsigned short* tir = &ti[wave][mt * 16 + lane][0];
                #pragma unroll 1
                for (int j = 0; j < TOK_TILE; ++j) {
                    float v = stage[wave][lane][j];
                    if (v > tvr[COARSE_K - 1]) {
                        int pos = COARSE_K - 1;
                        while (pos > 0 && tvr[pos - 1] < v) {
                            tvr[pos] = tvr[pos - 1];
                            tir[pos] = tir[pos - 1];
                            --pos;
                        }
                        tvr[pos] = v;
                        tir[pos] = (unsigned short)(n0 + j);
                    }
                }
            }
            __syncthreads();
        }
    }

    // merge 8 per-wave sorted lists -> per-split sorted top-20
    if (tid < TOKENS_BLK) {
        const int m = tid;
        int head[NWAVES];
        #pragma unroll
        for (int w = 0; w < NWAVES; ++w) head[w] = 0;
        const size_t base = ((size_t)(t0 + m) * NSPLIT + sp) * COARSE_K;
        for (int k = 0; k < COARSE_K; ++k) {
            float best = -3.4e38f; int bw = 0;
            #pragma unroll
            for (int w = 0; w < NWAVES; ++w) {
                float v = (head[w] < COARSE_K) ? tv[w][m][head[w]] : -3.4e38f;
                if (v > best) { best = v; bw = w; }
            }
            pval[base + k] = best;
            pidx[base + k] = (int)ti[bw][m][head[bw]];
            head[bw]++;
        }
    }
}

// merge NSPLIT sorted partial lists -> global sorted top-20 (lax.top_k order)
__global__ __launch_bounds__(NTHREADS)
void merge_coarse_kernel(const float* __restrict__ pval, const int* __restrict__ pidx,
                         int* __restrict__ out_cidx, float* __restrict__ out_cscore) {
    const int t = blockIdx.x * NTHREADS + threadIdx.x;
    if (t >= TOKENS_TOTAL) return;
    const float* pv = pval + (size_t)t * NSPLIT * COARSE_K;
    const int*   pi = pidx + (size_t)t * NSPLIT * COARSE_K;
    int head[NSPLIT];
    #pragma unroll
    for (int s = 0; s < NSPLIT; ++s) head[s] = 0;
    for (int k = 0; k < COARSE_K; ++k) {
        float best = -3.4e38f; int bs = 0;
        #pragma unroll
        for (int s = 0; s < NSPLIT; ++s) {
            float v = (head[s] < COARSE_K) ? pv[s * COARSE_K + head[s]] : -3.4e38f;
            if (v > best) { best = v; bs = s; }
        }
        out_cscore[(size_t)t * COARSE_K + k] = best;
        out_cidx[(size_t)t * COARSE_K + k]   = pi[bs * COARSE_K + head[bs]];
        head[bs]++;
    }
}

// ---------------------------------------------------------------------------
// Kernel: query WMMA (packed W_enc), fine scoring, top-10 + softmax, V gather
// ---------------------------------------------------------------------------
__global__ __launch_bounds__(NTHREADS)
void fine_stage_kernel(const unsigned short* __restrict__ xb,
                       const unsigned short* __restrict__ wep,
                       const float* __restrict__ K_all,
                       const float* __restrict__ V_all,
                       const int*   __restrict__ cidx,
                       float* __restrict__ out_output,
                       int*   __restrict__ out_fidx,
                       float* __restrict__ out_fw) {
    __shared__ __align__(16) unsigned short sx[TOK_TILE][D_MODEL];
    __shared__ float q[TOK_TILE][K_RANK];
    __shared__ float fs[TOK_TILE][COARSE_K];
    __shared__ int   gcand[TOK_TILE][COARSE_K];
    __shared__ float fwL[TOK_TILE][FINE_K];
    __shared__ int   fiL[TOK_TILE][FINE_K];

    const int t0   = blockIdx.x * TOK_TILE;
    const int tid  = threadIdx.x;
    const int wave = tid >> 5;
    const int lane = tid & 31;
    const int mrow = lane & 15;
    const int half = lane >> 4;

    {
        const u16x8* src = (const u16x8*)xb + (size_t)t0 * (D_MODEL / 8);
        u16x8* dst = (u16x8*)&sx[0][0];
        for (int i = tid; i < TOK_TILE * (D_MODEL / 8); i += NTHREADS) dst[i] = src[i];
    }
    __syncthreads();

    // query = x @ W_enc : [16 x 128]; wave w owns column tile w
    {
        const u16x16* bp = (const u16x16*)wep + (size_t)wave * (KSTEPS * 32);
        f32x8 c = {};
        for (int s = 0; s < KSTEPS; ++s) {
            u16x16 bv = bp[s * 32 + lane];
            bf16x16 b = __builtin_bit_cast(bf16x16, bv);
            const u16x8* row = (const u16x8*)&sx[mrow][0];
            u16x8 lo = row[(s << 2) + (half << 1)];
            u16x8 hi = row[(s << 2) + (half << 1) + 1];
            u16x16 av;
            #pragma unroll
            for (int j = 0; j < 8; ++j) { av[j] = lo[j]; av[j + 8] = hi[j]; }
            bf16x16 a = __builtin_bit_cast(bf16x16, av);
            c = __builtin_amdgcn_wmma_f32_16x16x32_bf16(
                    false, a, false, b, (short)0, c, false, false);
        }
        #pragma unroll
        for (int r = 0; r < 8; ++r)
            q[(half << 3) + r][wave * 16 + mrow] = c[r];
    }
    __syncthreads();

    // fine scores: one thread per (token, candidate) pair (320 pairs)
    if (tid < TOK_TILE * COARSE_K) {
        const int m  = tid / COARSE_K;
        const int cI = tid % COARSE_K;
        const int gi = cidx[(size_t)(t0 + m) * COARSE_K + cI];
        gcand[m][cI] = gi;
        const float* krow = K_all + (size_t)gi * K_RANK;
        float acc = 0.f;
        #pragma unroll 4
        for (int r = 0; r < K_RANK; ++r) acc += q[m][r] * krow[r];
        fs[m][cI] = acc * 0.08838834764831845f;  // 1/sqrt(128)
    }
    __syncthreads();

    // top-10 of 20 + softmax per token
    if (tid < TOK_TILE) {
        const int m = tid;
        unsigned int used = 0;
        float topv[FINE_K]; int topl[FINE_K];
        for (int f = 0; f < FINE_K; ++f) {
            float best = -3.4e38f; int bl = 0;
            for (int cI = 0; cI < COARSE_K; ++cI) {
                if (!((used >> cI) & 1u)) {
                    float v = fs[m][cI];
                    if (v > best) { best = v; bl = cI; }
                }
            }
            used |= 1u << bl;
            topv[f] = best; topl[f] = bl;
        }
        const float mx = topv[0];
        float e[FINE_K]; float sum = 0.f;
        for (int f = 0; f < FINE_K; ++f) { e[f] = __expf(topv[f] - mx); sum += e[f]; }
        const float inv = 1.f / sum;
        const size_t t = (size_t)(t0 + m);
        for (int f = 0; f < FINE_K; ++f) {
            const float wgt = e[f] * inv;
            const int   gi  = gcand[m][topl[f]];
            out_fidx[t * FINE_K + f] = gi;
            out_fw[t * FINE_K + f]   = wgt;
            fwL[m][f] = wgt;
            fiL[m][f] = gi;
        }
    }
    __syncthreads();

    // output[m][:] = sum_f w[m][f] * V_all[idx[m][f]][:]  (float4/thread sweep)
    const int d = tid * 4;  // 256 threads x 4 = 1024 columns
    for (int m = 0; m < TOK_TILE; ++m) {
        float4 acc = make_float4(0.f, 0.f, 0.f, 0.f);
        #pragma unroll
        for (int f = 0; f < FINE_K; ++f) {
            const float wgt = fwL[m][f];
            const float4 vv = *(const float4*)(V_all + (size_t)fiL[m][f] * D_MODEL + d);
            acc.x += wgt * vv.x; acc.y += wgt * vv.y;
            acc.z += wgt * vv.z; acc.w += wgt * vv.w;
        }
        *(float4*)(out_output + (size_t)(t0 + m) * D_MODEL + d) = acc;
    }
}

// ---------------------------------------------------------------------------
extern "C" void kernel_launch(void* const* d_in, const int* in_sizes, int n_in,
                              void* d_out, int out_size, void* d_ws, size_t ws_size,
                              hipStream_t stream) {
    (void)in_sizes; (void)n_in; (void)out_size; (void)ws_size;

    const float* x        = (const float*)d_in[0];
    const float* W_router = (const float*)d_in[1];
    const float* W_enc    = (const float*)d_in[2];
    const float* K_all    = (const float*)d_in[3];
    const float* V_all    = (const float*)d_in[4];

    // workspace layout (bytes)
    char* ws = (char*)d_ws;
    unsigned short* wrp  = (unsigned short*)ws;                              // 64 MB packed W_router
    unsigned short* xbf  = (unsigned short*)(ws + (size_t)67108864);         // 16 MB bf16 x
    unsigned short* wep  = (unsigned short*)(ws + (size_t)67108864 + 16777216);          // 256 KB packed W_enc
    float*          pval = (float*)(ws + (size_t)67108864 + 16777216 + 262144);          // 2.5 MB
    int*            pidx = (int*)(ws + (size_t)67108864 + 16777216 + 262144 + 2621440);  // 2.5 MB

    // output tuple, flat in return order
    float* out_f      = (float*)d_out;
    float* out_output = out_f;
    int*   out_cidx   = (int*)(out_f + (size_t)TOKENS_TOTAL * D_MODEL);
    float* out_cscore = out_f + (size_t)TOKENS_TOTAL * D_MODEL
                              + (size_t)TOKENS_TOTAL * COARSE_K;
    int*   out_fidx   = (int*)(out_f + (size_t)TOKENS_TOTAL * (D_MODEL + 2 * COARSE_K));
    float* out_fw     = out_f + (size_t)TOKENS_TOTAL * (D_MODEL + 2 * COARSE_K)
                              + (size_t)TOKENS_TOTAL * FINE_K;

    // 1) pack weights / activations to bf16 fragment layouts
    pack_w_kernel<<<dim3((N_KNOW / 16) * KSTEPS * 32 / NTHREADS), dim3(NTHREADS), 0, stream>>>(
        W_router, wrp, N_KNOW);
    pack_w_kernel<<<dim3((K_RANK / 16) * KSTEPS * 32 / NTHREADS), dim3(NTHREADS), 0, stream>>>(
        W_enc, wep, K_RANK);
    pack_x_kernel<<<dim3(TOKENS_TOTAL * D_MODEL / 8 / NTHREADS), dim3(NTHREADS), 0, stream>>>(
        x, xbf);

    // 2) coarse routing (WMMA) + per-split top-20, then merge
    coarse_partial_kernel<<<dim3((TOKENS_TOTAL / TOKENS_BLK) * NSPLIT), dim3(NTHREADS), 0, stream>>>(
        xbf, wrp, pval, pidx);
    merge_coarse_kernel<<<dim3(TOKENS_TOTAL / NTHREADS), dim3(NTHREADS), 0, stream>>>(
        pval, pidx, out_cidx, out_cscore);

    // 3) fine stage (WMMA query + scoring + top-10 softmax + V gather)
    fine_stage_kernel<<<dim3(TOKENS_TOTAL / TOK_TILE), dim3(NTHREADS), 0, stream>>>(
        xbf, wep, K_all, V_all, out_cidx, out_output, out_fidx, out_fw);
}
